// GRU_Cell_Zoneout_58050777972893
// MI455X (gfx1250) — matrix operand order repaired and checked by reading, not compiled, not run on hardware
//
#include <hip/hip_runtime.h>
#include <hip/hip_bf16.h>
#include <cstdint>

#define B_DIM   2048
#define IN_DIM  1024
#define H_DIM   1024
#define OUT_DIM 50257

typedef __attribute__((ext_vector_type(16))) __bf16 v16bf;
typedef __attribute__((ext_vector_type(8)))  float  v8f;
typedef __attribute__((ext_vector_type(4)))  float  f32x4;

union BF16Frag {
    v16bf        v;
    unsigned int u32[8];
};

// Pack two fp32 into one packed-bf16 dword (truncation) with a single v_perm_b32:
// out = { hi[31:24], hi[23:16], lo[31:24], lo[23:16] }
__device__ __forceinline__ unsigned int pack_bf16(float lo, float hi) {
    return __builtin_amdgcn_perm(__float_as_uint(hi), __float_as_uint(lo), 0x07060302u);
}

__device__ __forceinline__ void pack8(BF16Frag& f, int base, f32x4 a, f32x4 b) {
    f.u32[base + 0] = pack_bf16(a.x, a.y);
    f.u32[base + 1] = pack_bf16(a.z, a.w);
    f.u32[base + 2] = pack_bf16(b.x, b.y);
    f.u32[base + 3] = pack_bf16(b.z, b.w);
}

__device__ __forceinline__ int imin(int a, int b) { return a < b ? a : b; }

// Async 16-byte global -> LDS copy (CDNA5, ASYNCcnt-tracked).
// VDST carries the per-lane LDS byte address; 'off' = null SADDR (GV mode).
__device__ __forceinline__ void async_copy16(uint32_t lds_off, const float* gaddr) {
    asm volatile("global_load_async_to_lds_b128 %0, %1, off"
                 :
                 : "v"(lds_off), "v"((uint64_t)(uintptr_t)gaddr)
                 : "memory");
}

__device__ __forceinline__ uint32_t lds_addr(const void* p) {
    // generic LDS address carries the LDS byte offset in its low 32 bits
    return (uint32_t)(uintptr_t)p;
}

__device__ __forceinline__ void wait_async0() {
    asm volatile("s_wait_asynccnt 0" ::: "memory");
}

// ---------------------------------------------------------------------------
// Kernel 1: fused GRU gate GEMMs + zoneout -> ht
// 256 threads = 8 waves arranged 4(M) x 2(N); tile M=64, N=32.
// K loop 0..2047: k<1024 consumes (input, Wi*), k>=1024 (hidden, Wh*).
// Double-buffered async global->LDS staging.
// ---------------------------------------------------------------------------
__global__ __launch_bounds__(256) void gru_gates_kernel(
    const float* __restrict__ input, const float* __restrict__ hidden,
    const float* __restrict__ mask,
    const float* __restrict__ Wir, const float* __restrict__ bir,
    const float* __restrict__ Whr, const float* __restrict__ bhr,
    const float* __restrict__ Wiz, const float* __restrict__ biz,
    const float* __restrict__ Whz, const float* __restrict__ bhz,
    const float* __restrict__ Win, const float* __restrict__ bin,
    const float* __restrict__ Whn, const float* __restrict__ bhn,
    float* __restrict__ ht_out)
{
    __shared__ float As[2][64][32];      // A tile (rows of input/hidden), fp32
    __shared__ float Bs[2][3][32][32];   // r/z/n weight tiles [n][k], fp32

    const int tid   = threadIdx.x;
    const int lane  = tid & 31;
    const int wid   = tid >> 5;
    const int waveM = wid >> 1;             // 0..3
    const int waveN = wid & 1;              // 0..1
    const int blockM = blockIdx.x * 64;
    const int blockN = blockIdx.y * 32;

    // per-thread staging coordinates
    const int aRow0 = (tid + 0)   >> 3, aC0 = ((tid + 0)   & 7) * 4;
    const int aRow1 = (tid + 256) >> 3, aC1 = ((tid + 256) & 7) * 4;
    const int bRow  = tid >> 3,         bC  = (tid & 7) * 4;

    auto stage = [&](int k0, int buf) {
        const bool  ph0   = (k0 < IN_DIM);
        const float* Asrc = ph0 ? input : hidden;
        const float* Wr   = ph0 ? Wir : Whr;
        const float* Wz   = ph0 ? Wiz : Whz;
        const float* Wn   = ph0 ? Win : Whn;
        const int    kk   = k0 & (IN_DIM - 1);

        async_copy16(lds_addr(&As[buf][aRow0][aC0]),
                     Asrc + (size_t)(blockM + aRow0) * IN_DIM + kk + aC0);
        async_copy16(lds_addr(&As[buf][aRow1][aC1]),
                     Asrc + (size_t)(blockM + aRow1) * IN_DIM + kk + aC1);
        size_t wo = (size_t)(blockN + bRow) * IN_DIM + kk + bC;
        async_copy16(lds_addr(&Bs[buf][0][bRow][bC]), Wr + wo);
        async_copy16(lds_addr(&Bs[buf][1][bRow][bC]), Wz + wo);
        async_copy16(lds_addr(&Bs[buf][2][bRow][bC]), Wn + wo);
    };

    v8f acc_r = {}, acc_z = {}, acc_ni = {}, acc_nh = {};

    stage(0, 0);
    wait_async0();
    __syncthreads();

    int buf = 0;
    for (int k0 = 0; k0 < IN_DIM + H_DIM; k0 += 32) {
        if (k0 + 32 < IN_DIM + H_DIM) stage(k0 + 32, buf ^ 1);

        // A fragment (16x32 bf16): lanes 0-15 -> K[0..16); 16-31 -> K[16..32)
        BF16Frag af;
        {
            int row = waveM * 16 + (lane & 15);
            int kb  = (lane < 16) ? 0 : 16;
            const f32x4* p0 = (const f32x4*)&As[buf][row][kb];
            const f32x4* p1 = (const f32x4*)&As[buf][row][kb + 8];
            pack8(af, 0, p0[0], p0[1]);
            pack8(af, 4, p1[0], p1[1]);
        }
        // B fragments (32x16 bf16): lane -> column n, 16 contiguous K
        const int bn  = waveN * 16 + (lane & 15);
        const int bkb = (lane < 16) ? 0 : 16;
        BF16Frag bfr, bfz, bfn;
        {
            const f32x4* pr = (const f32x4*)&Bs[buf][0][bn][bkb];
            const f32x4* pz = (const f32x4*)&Bs[buf][1][bn][bkb];
            const f32x4* pn = (const f32x4*)&Bs[buf][2][bn][bkb];
            pack8(bfr, 0, pr[0], pr[1]); pack8(bfr, 4, pr[2], pr[3]);
            pack8(bfz, 0, pz[0], pz[1]); pack8(bfz, 4, pz[2], pz[3]);
            pack8(bfn, 0, pn[0], pn[1]); pack8(bfn, 4, pn[2], pn[3]);
        }

        acc_r = __builtin_amdgcn_wmma_f32_16x16x32_bf16(false, af.v, false, bfr.v,
                                                        (short)0, acc_r, false, false);
        acc_z = __builtin_amdgcn_wmma_f32_16x16x32_bf16(false, af.v, false, bfz.v,
                                                        (short)0, acc_z, false, false);
        if (k0 < IN_DIM)    // uniform branch
            acc_ni = __builtin_amdgcn_wmma_f32_16x16x32_bf16(false, af.v, false, bfn.v,
                                                             (short)0, acc_ni, false, false);
        else
            acc_nh = __builtin_amdgcn_wmma_f32_16x16x32_bf16(false, af.v, false, bfn.v,
                                                             (short)0, acc_nh, false, false);

        wait_async0();      // next-stage DMA has been hidden behind the WMMAs
        __syncthreads();
        buf ^= 1;
    }

    // Epilogue: C/D layout -> VGPR e holds M = e (+8 for lanes 16-31), N = lane&15
    const int n    = blockN + waveN * 16 + (lane & 15);
    const int mOff = (lane < 16) ? 0 : 8;
    const float b_r  = bir[n] + bhr[n];
    const float b_z  = biz[n] + bhz[n];
    const float b_ni = bin[n];
    const float b_nh = bhn[n];
    #pragma unroll
    for (int e = 0; e < 8; ++e) {
        int m = blockM + waveM * 16 + mOff + e;
        size_t o = (size_t)m * H_DIM + n;
        float r  = 1.0f / (1.0f + __expf(-(acc_r[e] + b_r)));
        float z  = 1.0f / (1.0f + __expf(-(acc_z[e] + b_z)));
        float nv = tanhf(acc_ni[e] + b_ni + r * (acc_nh[e] + b_nh));
        float mk = mask[o];
        float h0 = hidden[o];
        float zz = 1.0f - (1.0f - z) * mk;   // zoneout: force keep where mask==0
        ht_out[o] = (1.0f - zz) * nv + zz * h0;
    }
}

// ---------------------------------------------------------------------------
// Kernel 2: logits = ht @ h2o_w^T + b
// 256 threads = 8 waves arranged 2(M) x 4(N), each wave 32x32 (2x2 WMMA
// subtiles). Workgroup tile M=64, N=128. Double-buffered async staging.
// ---------------------------------------------------------------------------
__global__ __launch_bounds__(256) void gru_proj_kernel(
    const float* __restrict__ ht, const float* __restrict__ W,
    const float* __restrict__ bias, float* __restrict__ out)
{
    __shared__ float As[2][64][32];     // ht tile, fp32 (2 x 8 KB)
    __shared__ float Bs[2][128][32];    // W rows tile [n][k], fp32 (2 x 16 KB)

    const int tid   = threadIdx.x;
    const int lane  = tid & 31;
    const int wid   = tid >> 5;
    const int waveM = wid >> 2;             // 0..1
    const int waveN = wid & 3;              // 0..3
    const int blockM = blockIdx.y * 64;
    const int blockN = blockIdx.x * 128;

    const int aRow0 = (tid + 0)   >> 3, aC0 = ((tid + 0)   & 7) * 4;
    const int aRow1 = (tid + 256) >> 3, aC1 = ((tid + 256) & 7) * 4;

    auto stage = [&](int k0, int buf) {
        async_copy16(lds_addr(&As[buf][aRow0][aC0]),
                     ht + (size_t)(blockM + aRow0) * H_DIM + k0 + aC0);
        async_copy16(lds_addr(&As[buf][aRow1][aC1]),
                     ht + (size_t)(blockM + aRow1) * H_DIM + k0 + aC1);
        #pragma unroll
        for (int i = 0; i < 4; ++i) {
            int idx  = tid + i * 256;
            int row  = idx >> 3, c4 = (idx & 7) * 4;
            int wrow = imin(blockN + row, OUT_DIM - 1);   // clamp tail rows
            async_copy16(lds_addr(&Bs[buf][row][c4]),
                         W + (size_t)wrow * H_DIM + k0 + c4);
        }
    };

    v8f acc[2][2] = {};

    stage(0, 0);
    wait_async0();
    __syncthreads();

    int buf = 0;
    for (int k0 = 0; k0 < H_DIM; k0 += 32) {
        if (k0 + 32 < H_DIM) stage(k0 + 32, buf ^ 1);

        // prefetch the K-step after next of the weight tile into cache
        if (k0 + 64 < H_DIM) {
            int r = tid >> 1;
            __builtin_prefetch(W + (size_t)imin(blockN + r, OUT_DIM - 1) * H_DIM
                                 + (k0 + 64) + (tid & 1) * 16, 0, 0);
        }

        BF16Frag a[2], b[2];
        #pragma unroll
        for (int mi = 0; mi < 2; ++mi) {
            int row = waveM * 32 + mi * 16 + (lane & 15);
            int kb  = (lane < 16) ? 0 : 16;
            const f32x4* p0 = (const f32x4*)&As[buf][row][kb];
            const f32x4* p1 = (const f32x4*)&As[buf][row][kb + 8];
            pack8(a[mi], 0, p0[0], p0[1]);
            pack8(a[mi], 4, p1[0], p1[1]);
        }
        #pragma unroll
        for (int ni = 0; ni < 2; ++ni) {
            int row = waveN * 32 + ni * 16 + (lane & 15);
            int kb  = (lane < 16) ? 0 : 16;
            const f32x4* p = (const f32x4*)&Bs[buf][row][kb];
            pack8(b[ni], 0, p[0], p[1]);
            pack8(b[ni], 4, p[2], p[3]);
        }
        #pragma unroll
        for (int mi = 0; mi < 2; ++mi)
            #pragma unroll
            for (int ni = 0; ni < 2; ++ni)
                acc[mi][ni] = __builtin_amdgcn_wmma_f32_16x16x32_bf16(
                    false, a[mi].v, false, b[ni].v, (short)0, acc[mi][ni], false, false);

        wait_async0();
        __syncthreads();
        buf ^= 1;
    }

    const int mOff = (lane < 16) ? 0 : 8;
    #pragma unroll
    for (int mi = 0; mi < 2; ++mi) {
        #pragma unroll
        for (int ni = 0; ni < 2; ++ni) {
            int n = blockN + waveN * 32 + ni * 16 + (lane & 15);
            if (n < OUT_DIM) {
                float bv = bias[n];
                #pragma unroll
                for (int e = 0; e < 8; ++e) {
                    int m = blockM + waveM * 32 + mi * 16 + mOff + e;
                    out[(size_t)m * OUT_DIM + n] = acc[mi][ni][e] + bv;
                }
            }
        }
    }
}

// ---------------------------------------------------------------------------
// Kernel 3: in-place row-wise log_softmax over 50257 columns.
// Online max/logsumexp in one read pass + one subtract pass.
// ---------------------------------------------------------------------------
__global__ __launch_bounds__(256) void logsoftmax_kernel(float* __restrict__ out)
{
    __shared__ float redm[256];
    __shared__ float reds[256];

    float* p = out + (size_t)blockIdx.x * OUT_DIM;
    const int tid = threadIdx.x;

    float mx = -3.402823466e+38f;
    float sm = 0.0f;
    for (int i = tid; i < OUT_DIM; i += 256) {
        float x  = p[i];
        float m2 = fmaxf(mx, x);
        sm = sm * __expf(mx - m2) + __expf(x - m2);
        mx = m2;
    }
    redm[tid] = mx;
    reds[tid] = sm;
    __syncthreads();
    for (int s = 128; s > 0; s >>= 1) {
        if (tid < s) {
            float m1 = redm[tid], m2 = redm[tid + s];
            float mm = fmaxf(m1, m2);
            reds[tid] = reds[tid] * __expf(m1 - mm) + reds[tid + s] * __expf(m2 - mm);
            redm[tid] = mm;
        }
        __syncthreads();
    }
    const float lse = redm[0] + __logf(reds[0]);
    __syncthreads();

    for (int i = tid; i < OUT_DIM; i += 256)
        p[i] -= lse;
}

// ---------------------------------------------------------------------------
extern "C" void kernel_launch(void* const* d_in, const int* in_sizes, int n_in,
                              void* d_out, int out_size, void* d_ws, size_t ws_size,
                              hipStream_t stream) {
    (void)in_sizes; (void)n_in; (void)out_size; (void)d_ws; (void)ws_size;

    const float* input  = (const float*)d_in[0];
    const float* hidden = (const float*)d_in[1];
    const float* mask   = (const float*)d_in[2];
    const float* Wir_w  = (const float*)d_in[3];
    const float* Wir_b  = (const float*)d_in[4];
    const float* Whr_w  = (const float*)d_in[5];
    const float* Whr_b  = (const float*)d_in[6];
    const float* Wiz_w  = (const float*)d_in[7];
    const float* Wiz_b  = (const float*)d_in[8];
    const float* Whz_w  = (const float*)d_in[9];
    const float* Whz_b  = (const float*)d_in[10];
    const float* Win_w  = (const float*)d_in[11];
    const float* Win_b  = (const float*)d_in[12];
    const float* Whn_w  = (const float*)d_in[13];
    const float* Whn_b  = (const float*)d_in[14];
    const float* h2o_w  = (const float*)d_in[15];
    const float* h2o_b  = (const float*)d_in[16];

    float* out = (float*)d_out;                       // [B, OUT] log-probs
    float* ht  = out + (size_t)B_DIM * OUT_DIM;       // [B, H] new hidden

    gru_gates_kernel<<<dim3(B_DIM / 64, H_DIM / 32), 256, 0, stream>>>(
        input, hidden, mask,
        Wir_w, Wir_b, Whr_w, Whr_b,
        Wiz_w, Wiz_b, Whz_w, Whz_b,
        Win_w, Win_b, Whn_w, Whn_b, ht);

    gru_proj_kernel<<<dim3((OUT_DIM + 127) / 128, B_DIM / 64), 256, 0, stream>>>(
        ht, h2o_w, h2o_b, out);

    logsoftmax_kernel<<<dim3(B_DIM), 256, 0, stream>>>(out);
}
